// Deform_Conv2dBlock_23519240913140
// MI455X (gfx1250) — matrix operand order repaired and checked by reading, not compiled, not run on hardware
//
#include <hip/hip_runtime.h>
#include <hip/hip_bf16.h>

typedef _Float16 v8h  __attribute__((ext_vector_type(8)));
typedef _Float16 v16h __attribute__((ext_vector_type(16)));
typedef float    v8f  __attribute__((ext_vector_type(8)));

// Problem constants
#define B_   4
#define C_   128
#define H_   64
#define W_   64
#define OC_  128
#define HW_  (H_ * W_)          // 4096
#define NPIX (B_ * HW_)         // 16384
#define NTAP 9
#define KTOT (C_ * NTAP)        // 1152 ; reordered as K = t*128 + c
#define KSTEPS (KTOT / 32)      // 36
#define EPS_ 1e-5f

// ---------------------------------------------------------------------------
// Kernel 1: offset conv (3x3, C=128 -> 18) + per-(pixel,tap) bilinear metadata
// ---------------------------------------------------------------------------
__global__ void offset_meta_kernel(const float* __restrict__ x,
                                   const float* __restrict__ pw,   // (18,128,3,3)
                                   const float* __restrict__ pb,   // (18,)
                                   int4*  __restrict__ mi,         // (NPIX*9) gather idx (hw)
                                   float4* __restrict__ mf)        // (NPIX*9) bilinear w
{
    int p = blockIdx.x * blockDim.x + threadIdx.x;
    if (p >= NPIX) return;
    int b = p >> 12, hw = p & (HW_ - 1);
    int h = hw >> 6, w = hw & 63;

    float acc[18];
#pragma unroll
    for (int o = 0; o < 18; ++o) acc[o] = pb[o];

    const float* xb = x + ((long)b * C_ * HW_);
    for (int c = 0; c < C_; ++c) {
        const float* xc = xb + (c << 12);
        float xv[9];
#pragma unroll
        for (int kh = 0; kh < 3; ++kh)
#pragma unroll
            for (int kw = 0; kw < 3; ++kw) {
                int hh = h - 1 + kh, ww = w - 1 + kw;
                xv[kh * 3 + kw] = (hh >= 0 && hh < H_ && ww >= 0 && ww < W_)
                                      ? xc[(hh << 6) + ww] : 0.f;
            }
#pragma unroll
        for (int o = 0; o < 18; ++o) {
            const float* wp = pw + ((o * C_ + c) * 9);
            float a = 0.f;
#pragma unroll
            for (int t = 0; t < 9; ++t) a += xv[t] * wp[t];
            acc[o] += a;
        }
    }

    // lr_ratio trick is forward-identity; build sampling metadata.
    const float HPm1 = (float)(H_ + 1);   // 65 = Hp-1
    const float WPm1 = (float)(W_ + 1);
    for (int t = 0; t < 9; ++t) {
        float px = acc[t]     + (float)(h + 1) + (float)(t / 3 - 1);
        float py = acc[t + 9] + (float)(w + 1) + (float)(t % 3 - 1);
        float fx = floorf(px), fy = floorf(py);
        float ltx = fminf(fmaxf(fx, 0.f), HPm1);
        float lty = fminf(fmaxf(fy, 0.f), WPm1);
        float rbx = fminf(fmaxf(fx + 1.f, 0.f), HPm1);
        float rby = fminf(fmaxf(fy + 1.f, 0.f), WPm1);
        float pxc = fminf(fmaxf(px, 0.f), HPm1);
        float pyc = fminf(fmaxf(py, 0.f), WPm1);

        float g_lt = (1.f + (ltx - pxc)) * (1.f + (lty - pyc));
        float g_rb = (1.f - (rbx - pxc)) * (1.f - (rby - pyc));
        float g_lb = (1.f + (ltx - pxc)) * (1.f - (rby - pyc));
        float g_rt = (1.f - (rbx - pxc)) * (1.f + (lty - pyc));

        int iltx = (int)ltx, ilty = (int)lty, irbx = (int)rbx, irby = (int)rby;
        // fold zero-pad border (coord 0 or 65 in padded frame) into weight=0
        int4 qi; float4 qw;
        bool z;
        z = (iltx < 1 || iltx > H_ || ilty < 1 || ilty > W_);
        qi.x = z ? 0 : ((iltx - 1) << 6) + (ilty - 1);  qw.x = z ? 0.f : g_lt;
        z = (irbx < 1 || irbx > H_ || irby < 1 || irby > W_);
        qi.y = z ? 0 : ((irbx - 1) << 6) + (irby - 1);  qw.y = z ? 0.f : g_rb;
        z = (iltx < 1 || iltx > H_ || irby < 1 || irby > W_);
        qi.z = z ? 0 : ((iltx - 1) << 6) + (irby - 1);  qw.z = z ? 0.f : g_lb;
        z = (irbx < 1 || irbx > H_ || ilty < 1 || ilty > W_);
        qi.w = z ? 0 : ((irbx - 1) << 6) + (ilty - 1);  qw.w = z ? 0.f : g_rt;

        long r = (long)p * 9 + t;
        mi[r] = qi;
        mf[r] = qw;
    }
}

// ---------------------------------------------------------------------------
// Kernel 2a: NCHW -> NHWC copy of x (coalesced reads, L2-resident writes)
// ---------------------------------------------------------------------------
__global__ void nhwc_kernel(const float* __restrict__ x, float* __restrict__ xn)
{
    long i = (long)blockIdx.x * blockDim.x + threadIdx.x;   // (b,c,hw) order
    if (i >= (long)B_ * C_ * HW_) return;
    int hw = (int)(i & (HW_ - 1));
    int c  = (int)((i >> 12) & (C_ - 1));
    int b  = (int)(i >> 19);
    xn[(((long)b << 12) + hw) * C_ + c] = x[i];
}

// ---------------------------------------------------------------------------
// Kernel 2b: reorder main-conv weights to f16 [OC][K = t*128 + c]; zero stats
// ---------------------------------------------------------------------------
__global__ void prep_kernel(const float* __restrict__ conv_w,   // (OC,C,9)
                            _Float16* __restrict__ w16,
                            float* __restrict__ stats)          // 256 floats
{
    int i = blockIdx.x * blockDim.x + threadIdx.x;
    if (i < 256) stats[i] = 0.f;
    if (i < OC_ * KTOT) {
        int oc  = i / KTOT;
        int rem = i - oc * KTOT;     // source index c*9 + t
        int c   = rem / 9;
        int t   = rem - c * 9;
        w16[oc * KTOT + t * C_ + c] = (_Float16)conv_w[i];
    }
}

// ---------------------------------------------------------------------------
// Kernel 3: deformable-sample + WMMA GEMM ; D[128 oc x 32 pix] per block
//   K reordered tap-major: each 32-K slice = one tap, 32 contiguous channels.
// ---------------------------------------------------------------------------
__global__ void __launch_bounds__(256)
deform_wmma_kernel(const float* __restrict__ xn,      // NHWC
                   const _Float16* __restrict__ w16,  // [OC][1152] tap-major K
                   const int4*   __restrict__ mi,
                   const float4* __restrict__ mf,
                   float* __restrict__ out_pre,
                   float* __restrict__ stat_sum,
                   float* __restrict__ stat_sq)
{
    // [K-slice u of pair][N sub-tile][B fragment, contiguous K order]
    // lane L holds halves j=0..15 with K=(L/16)*16+j, column n=L%16.
    __shared__ __align__(32) _Float16 sbuf[2][2][32][16];
    __shared__ int4   smi[32 * NTAP];     // per-block meta: 32 pixels x 9 taps
    __shared__ float4 smf[32 * NTAP];

    const int tile = blockIdx.x;          // 512 tiles of 32 pixels
    const int P0   = tile * 32;
    const int b    = P0 >> 12;
    const int hw0  = P0 & (HW_ - 1);
    const int tid  = threadIdx.x;
    const int lane = tid & 31;
    const int wave = tid >> 5;

    // hoist the 288 contiguous meta records for this block into LDS
    for (int idx = tid; idx < 32 * NTAP; idx += 256) {
        long r = (long)P0 * NTAP + idx;
        smi[idx] = mi[r];
        smf[idx] = mf[r];
    }
    __syncthreads();

    const int mrow  = wave * 16 + (lane & 15);   // A row (output channel)
    const int kgrpA = (lane >> 4) * 16;          // A layout K group per half-wave

    const float* xb = xn + ((long)b << 12) * C_; // batch base in NHWC

    // per-thread staging coordinates: 4 samples/step, coalesced over channels
    const int c_l  = tid & 31;                   // channel within 32-slice
    const int pixb = tid >> 5;                   // base pixel 0..7
    const int Lrow = ((c_l >> 4) << 4);          // LDS row group from K
    const int jcol = c_l & 15;

    v8f acc0 = {};
    v8f acc1 = {};

    for (int ks = 0; ks < KSTEPS; ks += 2) {
#pragma unroll
        for (int u = 0; u < 2; ++u) {
            const int kss   = ks + u;
            const int t     = kss >> 2;          // tap for this K-slice
            const int c     = ((kss & 3) << 5) + c_l;
            const float* xc = xb + c;
#pragma unroll
            for (int pp = 0; pp < 4; ++pp) {
                const int pix = pixb + pp * 8;   // 0..31
                int4   qi = smi[pix * NTAP + t];
                float4 qw = smf[pix * NTAP + t];
                float v = qw.x * xc[(long)qi.x << 7] + qw.y * xc[(long)qi.y << 7]
                        + qw.z * xc[(long)qi.z << 7] + qw.w * xc[(long)qi.w << 7];
                sbuf[u][pix >> 4][Lrow + (pix & 15)][jcol] = (_Float16)v;
            }
        }
        __syncthreads();

#pragma unroll
        for (int u = 0; u < 2; ++u) {
            // A fragment (weights) in contiguous K order 16*(lane/16)+s, as B
            const _Float16* wa = w16 + (long)mrow * KTOT + (ks + u) * 32 + kgrpA;
            v8h lo = *(const v8h*)(wa);
            v8h hi = *(const v8h*)(wa + 8);
            v16h afrag = __builtin_shufflevector(lo, hi,
                             0, 1, 2, 3, 4, 5, 6, 7, 8, 9, 10, 11, 12, 13, 14, 15);
            // B fragments straight from LDS (contiguous 32B per lane)
            v16h b0 = *(const v16h*)(&sbuf[u][0][lane][0]);
            v16h b1 = *(const v16h*)(&sbuf[u][1][lane][0]);
            acc0 = __builtin_amdgcn_wmma_f32_16x16x32_f16(
                       false, afrag, false, b0, (short)0, acc0, false, false);
            acc1 = __builtin_amdgcn_wmma_f32_16x16x32_f16(
                       false, afrag, false, b1, (short)0, acc1, false, false);
        }
        __syncthreads();
    }

    // epilogue: D[r + 8*(lane/16)][lane%16], write pre-BN + BN stats
    const int n      = lane & 15;
    const int chbase = wave * 16 + ((lane >> 4) << 3);
#pragma unroll
    for (int r = 0; r < 8; ++r) {
        float v0 = acc0[r];
        float v1 = acc1[r];
        long base = (((long)b * OC_ + chbase + r) << 12) + hw0 + n;
        out_pre[base]      = v0;
        out_pre[base + 16] = v1;
        float s = v0 + v1, q = v0 * v0 + v1 * v1;
#pragma unroll
        for (int m = 1; m < 16; m <<= 1) {
            s += __shfl_xor(s, m, 16);
            q += __shfl_xor(q, m, 16);
        }
        if (n == 0) {
            atomicAdd(&stat_sum[chbase + r], s);
            atomicAdd(&stat_sq [chbase + r], q);
        }
    }
}

// ---------------------------------------------------------------------------
// Kernel 4: BatchNorm (batch stats) + ReLU
// ---------------------------------------------------------------------------
__global__ void bn_relu_kernel(const float* __restrict__ pre,
                               const float* __restrict__ ssum,
                               const float* __restrict__ ssq,
                               const float* __restrict__ gamma,
                               const float* __restrict__ beta,
                               float* __restrict__ out)
{
    long i = (long)blockIdx.x * blockDim.x + threadIdx.x;
    if (i >= (long)B_ * OC_ * HW_) return;
    int c = (int)((i >> 12) & (OC_ - 1));
    const float inv_n = 1.f / (float)NPIX;
    float mean = ssum[c] * inv_n;
    float var  = ssq[c] * inv_n - mean * mean;
    float v    = pre[i];
    float o    = gamma[c] * (v - mean) * rsqrtf(var + EPS_) + beta[c];
    out[i] = fmaxf(o, 0.f);
}

// ---------------------------------------------------------------------------
extern "C" void kernel_launch(void* const* d_in, const int* in_sizes, int n_in,
                              void* d_out, int out_size, void* d_ws, size_t ws_size,
                              hipStream_t stream)
{
    const float* x        = (const float*)d_in[0];
    const float* p_conv_w = (const float*)d_in[1];
    const float* p_conv_b = (const float*)d_in[2];
    const float* conv_w   = (const float*)d_in[3];
    const float* gamma    = (const float*)d_in[4];
    const float* beta     = (const float*)d_in[5];
    float* out = (float*)d_out;

    char* ws = (char*)d_ws;
    // workspace layout (all 16B-aligned offsets)
    float*    stats = (float*)ws;                               // 256 f32 (sum|sq)
    _Float16* w16   = (_Float16*)(ws + 1024);                   // 128*1152 f16
    size_t off = 1024 + (size_t)OC_ * KTOT * sizeof(_Float16);  // 295936
    int4*   mi = (int4*)(ws + off);                             // NPIX*9 int4
    off += (size_t)NPIX * NTAP * sizeof(int4);
    float4* mf = (float4*)(ws + off);                           // NPIX*9 float4
    off += (size_t)NPIX * NTAP * sizeof(float4);
    float* xn = (float*)(ws + off);                             // NHWC copy of x
    off += (size_t)B_ * C_ * HW_ * sizeof(float);
    float* out_pre = (float*)(ws + off);                        // B*OC*HW f32

    float* stat_sum = stats;
    float* stat_sq  = stats + 128;

    // 1) offsets + bilinear metadata
    offset_meta_kernel<<<NPIX / 256, 256, 0, stream>>>(x, p_conv_w, p_conv_b, mi, mf);
    // 2) NHWC transpose + f16 tap-major weights + zero stats
    nhwc_kernel<<<(B_ * C_ * HW_ + 255) / 256, 256, 0, stream>>>(x, xn);
    prep_kernel<<<(OC_ * KTOT + 255) / 256, 256, 0, stream>>>(conv_w, w16, stats);
    // 3) WMMA GEMM over reordered (t,c) contraction, 32-pixel N tiles
    deform_wmma_kernel<<<NPIX / 32, 256, 0, stream>>>(xn, w16, mi, mf, out_pre,
                                                      stat_sum, stat_sq);
    // 4) BN + ReLU
    bn_relu_kernel<<<((long)B_ * OC_ * HW_ + 255) / 256, 256, 0, stream>>>(
        out_pre, stat_sum, stat_sq, gamma, beta, out);
}
